// DiMPSteepestDescentGN_58798102282932
// MI455X (gfx1250) — compile-verified, matrix operand order checked by reading, not run on hardware
//
#include <hip/hip_runtime.h>
#include <hip/hip_bf16.h>
#include <cstdint>

// DiMP steepest-descent GN optimizer for MI455X (gfx1250, wave32).
// Heavy correlations cast as per-s GEMMs:
//   GEMM1/3 (reduce over C):  bf16 feat copy [s][m][c]  x  wB [s][n][c]   -> v_wmma_f32_16x16x32_bf16
//                             B matrix staged per-block into LDS via global_load_async_to_lds_b128
//   GEMM2   (reduce over m):  original fp32 feat (c-major) x resT [s][n][m] -> v_wmma_f32_16x16x4_f32
// Bandwidth-bound: 11 bf16 feat passes + 5 fp32 feat passes ~= 5.3 GB @ 23.3 TB/s.

#define I_N   8
#define S_N   64
#define C_N   512
#define H_N   22
#define W_N   22
#define HW_N  484
#define M_N   3872      // I_N * HW_N
#define O_N   23
#define OO_N  529
#define F_N   16        // fh*fw taps
#define NITER 5
#define WELEM 524288    // S_N*C_N*F_N
#define MTILES 242      // M_N / 16

typedef __bf16 bf16_t;
typedef bf16_t v16bf __attribute__((ext_vector_type(16)));
typedef bf16_t v8bf  __attribute__((ext_vector_type(8)));
typedef float  v8f   __attribute__((ext_vector_type(8)));
typedef float  v2f   __attribute__((ext_vector_type(2)));

__device__ __forceinline__ float blockReduce256(float v) {
  __shared__ float red[256];
  int tid = threadIdx.x;
  red[tid] = v;
  __syncthreads();
  #pragma unroll
  for (int off = 128; off > 0; off >>= 1) {
    if (tid < off) red[tid] += red[tid + off];
    __syncthreads();
  }
  return red[0];
}

// feat f32 [i][s][c][y,x]  ->  featA bf16 [s][m=i*484+yx][c]   (LDS transpose tile 32c x 32m)
__global__ __launch_bounds__(256) void prep_feat(const float* __restrict__ feat,
                                                 bf16_t* __restrict__ featA) {
  int si = blockIdx.y; int s = si >> 3; int i = si & 7;
  int bt = blockIdx.x; int ct = bt >> 4; int mt = bt & 15;
  int c0 = ct * 32, m0 = mt * 32;
  __shared__ float tile[32][33];
  int tid = threadIdx.x;
  int ml = tid & 31, cg = tid >> 5;
  for (int cc = cg; cc < 32; cc += 8) {
    int m = m0 + ml;
    float v = 0.0f;
    if (m < HW_N) v = feat[(((size_t)i * S_N + s) * C_N + c0 + cc) * HW_N + m];
    tile[cc][ml] = v;
  }
  __syncthreads();
  int cl = tid & 31, mg = tid >> 5;
  for (int mm = mg; mm < 32; mm += 8) {
    int m = m0 + mm;
    if (m < HW_N)
      featA[((size_t)s * M_N + (size_t)i * HW_N + m) * C_N + c0 + cl] = (bf16_t)tile[cl][mm];
  }
}

// label / sigmoid(mask) / sample-weight maps from bb; also zeroes the 6 loss slots.
__global__ __launch_bounds__(256) void prep_maps(const float* __restrict__ bb,
    const float* __restrict__ label_w, const float* __restrict__ mask_w,
    const float* __restrict__ spatial_w, float* __restrict__ label,
    float* __restrict__ mask, float* __restrict__ swb, float* __restrict__ lossd) {
  int s = blockIdx.y;
  if (blockIdx.y == 0 && blockIdx.x == 0 && threadIdx.x < NITER + 1) lossd[threadIdx.x] = 0.0f;
  int t = blockIdx.x * 256 + threadIdx.x;
  if (t >= I_N * OO_N) return;
  int i = t / OO_N, p = t % OO_N;
  int oy = p / O_N, ox = p % O_N;
  const float* b4 = bb + ((size_t)i * S_N + s) * 4;
  float cr = (b4[1] + 0.5f * b4[3]) * (1.0f / 16.0f);   // row center (y)
  float cc = (b4[0] + 0.5f * b4[2]) * (1.0f / 16.0f);   // col center (x)
  float d0 = (float)oy - cr, d1 = (float)ox - cc;
  float dist = sqrtf(d0 * d0 + d1 * d1);
  float lab = 0.0f, mk = 0.0f, sp = 0.0f;
  #pragma unroll
  for (int b = 0; b < 4; ++b) {
    float v = fmaxf(1.0f - fabsf(dist - (float)b), 0.0f);
    lab += v * label_w[b]; mk += v * mask_w[b]; sp += v * spatial_w[b];
  }
  float v4 = fminf(fmaxf(dist - 3.0f, 0.0f), 1.0f);
  lab += v4 * label_w[4]; mk += v4 * mask_w[4]; sp += v4 * spatial_w[4];
  size_t idx = ((size_t)s * I_N + i) * OO_N + p;
  label[idx] = lab;
  mask[idx]  = 1.0f / (1.0f + expf(-mk));
  swb[idx]   = sp * 0.35355339059327373f;               // sqrt(1/I), I=8
}

// wcur = weights; iterates[0] = weights; wB bf16 [s][n][c] (transposed for B fragments)
__global__ __launch_bounds__(256) void prep_w(const float* __restrict__ w0,
    float* __restrict__ wcur, bf16_t* __restrict__ wB, float* __restrict__ out) {
  int idx = blockIdx.x * 256 + threadIdx.x;
  int s = idx >> 13; int rem = idx & 8191; int c = rem >> 4; int n = rem & 15;
  float w = w0[idx];
  wcur[idx] = w;
  out[(size_t)WELEM + idx] = w;
  wB[((size_t)s * F_N + n) * C_N + c] = (bf16_t)w;
}

// partial[s][m][n] = sum_c featA[s][m][c] * B[s][n][c]  -- bf16 WMMA.
// B (16 KB, shared by all 4 waves of the block) staged into LDS with async loads.
__global__ __launch_bounds__(128) void gemm_mn(const bf16_t* __restrict__ featA,
                                               const bf16_t* __restrict__ Bmat,
                                               float* __restrict__ partial) {
  int s = blockIdx.y;
  __shared__ bf16_t wLDS[F_N * C_N];            // 16 KB
  {
    uint32_t ldsBase = (uint32_t)(uintptr_t)&wLDS[0];   // low 32 bits == LDS byte offset
    uint64_t gBase   = (uint64_t)(uintptr_t)(Bmat + (size_t)s * F_N * C_N);
    for (int ofs = threadIdx.x * 16; ofs < F_N * C_N * 2; ofs += 128 * 16) {
      uint32_t la = ldsBase + (uint32_t)ofs;
      uint64_t ga = gBase + (uint64_t)ofs;
      asm volatile("global_load_async_to_lds_b128 %0, %1, off"
                   :: "v"(la), "v"(ga) : "memory");
    }
    asm volatile("s_wait_asynccnt 0x0" ::: "memory");   // own wave's async loads done
  }
  __syncthreads();                                      // all waves' B staged
  int wave = threadIdx.x >> 5, lane = threadIdx.x & 31;
  int mt = blockIdx.x * 4 + wave;
  if (mt >= MTILES) return;                 // wave-uniform exit: EXEC all-ones for WMMA
  int m0 = mt * 16;
  int row = lane & 15;
  int hi  = lane >> 4;                      // K-half select per A/B 16-bit layout
  const bf16_t* aBase = featA + ((size_t)s * M_N + m0 + row) * C_N + hi * 8;
  const bf16_t* bBase = &wLDS[row * C_N + hi * 8];
  v8f acc = {};
  #pragma unroll 4
  for (int c0 = 0; c0 < C_N; c0 += 32) {
    __builtin_prefetch(aBase + c0 + 128, 0, 1);
    v8bf alo = *(const v8bf*)(aBase + c0);
    v8bf ahi = *(const v8bf*)(aBase + c0 + 16);
    v8bf blo = *(const v8bf*)(bBase + c0);          // ds_load_b128 from LDS
    v8bf bhi = *(const v8bf*)(bBase + c0 + 16);
    v16bf a = __builtin_shufflevector(alo, ahi, 0,1,2,3,4,5,6,7,8,9,10,11,12,13,14,15);
    v16bf b = __builtin_shufflevector(blo, bhi, 0,1,2,3,4,5,6,7,8,9,10,11,12,13,14,15);
    acc = __builtin_amdgcn_wmma_f32_16x16x32_bf16(false, a, false, b, (short)0, acc,
                                                  false, false);
  }
  float* outp = partial + ((size_t)s * M_N + m0 + hi * 8) * F_N + row;
  #pragma unroll
  for (int r = 0; r < 8; ++r) outp[(size_t)r * F_N] = acc[r];
}

// scores gather + LeakyReluPar residual; loss atomic; stores s_der and res_mapped
__global__ __launch_bounds__(256) void scores_resid(const float* __restrict__ partial,
    const float* __restrict__ label, const float* __restrict__ mask,
    const float* __restrict__ swb, float* __restrict__ sder, float* __restrict__ rm,
    float* __restrict__ loss_slot, int store) {
  int s = blockIdx.y;
  int t = blockIdx.x * 256 + threadIdx.x;
  float lsum = 0.0f;
  if (t < I_N * OO_N) {
    int i = t / OO_N, p = t % OO_N;
    int oy = p / O_N, ox = p % O_N;
    float sc = 0.0f;
    #pragma unroll
    for (int fh = 0; fh < 4; ++fh) {
      int y = oy + fh - 2;
      if ((unsigned)y < (unsigned)H_N) {
        #pragma unroll
        for (int fw = 0; fw < 4; ++fw) {
          int x = ox + fw - 2;
          if ((unsigned)x < (unsigned)W_N)
            sc += partial[((size_t)s * M_N + i * HW_N + y * W_N + x) * F_N + fh * 4 + fw];
        }
      }
    }
    size_t idx = ((size_t)s * I_N + i) * OO_N + p;
    float mk = mask[idx], lb = label[idx], w = swb[idx];
    float sgn = (sc > 0.0f) ? 1.0f : ((sc < 0.0f) ? -1.0f : 0.0f);
    float act = 0.5f * (1.0f - mk) * fabsf(sc) + 0.5f * (1.0f + mk) * sc;
    float der = 0.5f * (1.0f - mk) * sgn + 0.5f * (1.0f + mk);
    float res = w * (act - lb);
    if (store) { sder[idx] = der; rm[idx] = der * w * res; }
    lsum = res * res;
  }
  float tot = blockReduce256(lsum);
  if (threadIdx.x == 0) atomicAdd(loss_slot, tot * (1.0f / S_N));
}

__global__ __launch_bounds__(256) void reg_loss(const float* __restrict__ wcur,
    const float* __restrict__ freg, float* __restrict__ loss_slot) {
  int idx = blockIdx.x * 256 + threadIdx.x;
  float fr = freg[0];
  float reg = fmaxf(fr * fr, 1e-6f);
  float w = wcur[idx];
  float tot = blockReduce256(w * w);
  if (threadIdx.x == 0) atomicAdd(loss_slot, tot * reg * (1.0f / S_N));
}

// resT[s][n][m] = res_mapped shifted (zero padded); also zero a_num/a_den for this iter
__global__ __launch_bounds__(256) void shift_res(const float* __restrict__ rm,
    float* __restrict__ resT, float* __restrict__ anum, float* __restrict__ aden) {
  int s = blockIdx.y;
  if (blockIdx.x == 0 && threadIdx.x == 0) { anum[s] = 0.0f; aden[s] = 0.0f; }
  int t = blockIdx.x * 256 + threadIdx.x;        // 16*3872 = 61952 = 242*256 exact
  int n = t / M_N, m = t % M_N;
  int fh = n >> 2, fw = n & 3;
  int i = m / HW_N, yx = m % HW_N;
  int y = yx / W_N, x = yx % W_N;
  int oy = y - fh + 2, ox = x - fw + 2;
  float v = 0.0f;
  if ((unsigned)oy < (unsigned)O_N && (unsigned)ox < (unsigned)O_N)
    v = rm[((size_t)s * I_N + i) * OO_N + oy * O_N + ox];
  resT[((size_t)s * F_N + n) * M_N + m] = v;
}

// w_grad[c][n] = sum_m feat[m,c]*resT[n,m] + reg*w  -- fp32 WMMA on original c-major feat
__global__ __launch_bounds__(128) void gemm_grad(const float* __restrict__ feat,
    const float* __restrict__ resT, const float* __restrict__ wcur,
    const float* __restrict__ freg, float* __restrict__ wgrad,
    bf16_t* __restrict__ wgradB, float* __restrict__ anum) {
  int s = blockIdx.y;
  int wave = threadIdx.x >> 5, lane = threadIdx.x & 31;
  int ct = blockIdx.x * 4 + wave;               // 32 c-tiles, grid.x = 8
  int c0 = ct * 16;
  int row = lane & 15, hi = lane >> 4;
  v8f acc = {};
  for (int i = 0; i < I_N; ++i) {
    const float* aI = feat + (((size_t)i * S_N + s) * C_N + c0 + row) * HW_N + hi * 2;
    const float* bI = resT + ((size_t)s * F_N + row) * M_N + i * HW_N + hi * 2;
    #pragma unroll 4
    for (int yx = 0; yx < HW_N; yx += 4) {      // 484 % 4 == 0, never crosses i
      v2f a = *(const v2f*)(aI + yx);
      v2f b = *(const v2f*)(bI + yx);
      acc = __builtin_amdgcn_wmma_f32_16x16x4_f32(false, a, false, b, (short)0, acc,
                                                  false, false);
    }
  }
  float fr = freg[0];
  float reg = fmaxf(fr * fr, 1e-6f);
  float lsum = 0.0f;
  #pragma unroll
  for (int r = 0; r < 8; ++r) {
    int c = c0 + r + hi * 8;
    size_t wi = ((size_t)s * C_N + c) * F_N + row;
    float wg = acc[r] + reg * wcur[wi];
    wgrad[wi] = wg;
    wgradB[((size_t)s * F_N + row) * C_N + c] = (bf16_t)wg;
    lsum += wg * wg;
  }
  atomicAdd(&anum[s], lsum);
}

// sg = sw * s_der * conv(feat, w_grad); accumulate a_den[s] += sg^2
__global__ __launch_bounds__(256) void sg_reduce(const float* __restrict__ partial2,
    const float* __restrict__ swb, const float* __restrict__ sder,
    float* __restrict__ aden) {
  int s = blockIdx.y;
  int t = blockIdx.x * 256 + threadIdx.x;
  float lsum = 0.0f;
  if (t < I_N * OO_N) {
    int i = t / OO_N, p = t % OO_N;
    int oy = p / O_N, ox = p % O_N;
    float sc = 0.0f;
    #pragma unroll
    for (int fh = 0; fh < 4; ++fh) {
      int y = oy + fh - 2;
      if ((unsigned)y < (unsigned)H_N) {
        #pragma unroll
        for (int fw = 0; fw < 4; ++fw) {
          int x = ox + fw - 2;
          if ((unsigned)x < (unsigned)W_N)
            sc += partial2[((size_t)s * M_N + i * HW_N + y * W_N + x) * F_N + fh * 4 + fw];
        }
      }
    }
    size_t idx = ((size_t)s * I_N + i) * OO_N + p;
    float sgv = swb[idx] * sder[idx] * sc;
    lsum = sgv * sgv;
  }
  float tot = blockReduce256(lsum);
  if (threadIdx.x == 0) atomicAdd(&aden[s], tot);
}

__global__ __launch_bounds__(256) void update_w(const float* __restrict__ wgrad,
    const float* __restrict__ anum, const float* __restrict__ aden,
    const float* __restrict__ lsl, const float* __restrict__ freg,
    float* __restrict__ wcur, bf16_t* __restrict__ wB, float* __restrict__ out, int t) {
  int idx = blockIdx.x * 256 + threadIdx.x;
  int s = idx >> 13; int rem = idx & 8191; int c = rem >> 4; int n = rem & 15;
  float fr = freg[0];
  float reg = fmaxf(fr * fr, 1e-6f);
  float step = expf(lsl[0]);
  float an = anum[s];
  float ad = fmaxf(aden[s] + reg * an, 1e-8f);
  float alpha = an / ad;
  float wnew = wcur[idx] - step * alpha * wgrad[idx];
  wcur[idx] = wnew;
  wB[((size_t)s * F_N + n) * C_N + c] = (bf16_t)wnew;
  out[(size_t)WELEM * (t + 2) + idx] = wnew;          // iterates[t+1]
  if (t == NITER - 1) out[idx] = wnew;                // final weights
}

extern "C" void kernel_launch(void* const* d_in, const int* in_sizes, int n_in,
                              void* d_out, int out_size, void* d_ws, size_t ws_size,
                              hipStream_t stream) {
  (void)in_sizes; (void)n_in; (void)out_size; (void)ws_size;
  const float* weights = (const float*)d_in[0];
  const float* feat    = (const float*)d_in[1];
  const float* bb      = (const float*)d_in[2];
  const float* lsl     = (const float*)d_in[3];
  const float* freg    = (const float*)d_in[4];
  const float* label_w = (const float*)d_in[5];
  const float* mask_w  = (const float*)d_in[6];
  const float* spat_w  = (const float*)d_in[7];
  float* out   = (float*)d_out;
  float* lossd = out + (size_t)WELEM * 7;             // 6 loss slots after weights+iterates

  char* ws = (char*)d_ws;
  size_t off = 0;
  auto take = [&](size_t bytes) {
    char* p = ws + off; off += (bytes + 255) & ~(size_t)255; return p;
  };
  bf16_t* featA  = (bf16_t*)take((size_t)S_N * M_N * C_N * 2);   // 253.8 MB
  float*  wcur   = (float*) take((size_t)WELEM * 4);
  bf16_t* wB     = (bf16_t*)take((size_t)WELEM * 2);
  float*  wgrad  = (float*) take((size_t)WELEM * 4);
  bf16_t* wgradB = (bf16_t*)take((size_t)WELEM * 2);
  float*  part1  = (float*) take((size_t)S_N * M_N * F_N * 4);
  float*  part2  = (float*) take((size_t)S_N * M_N * F_N * 4);
  float*  resT   = (float*) take((size_t)S_N * F_N * M_N * 4);
  float*  labelb = (float*) take((size_t)S_N * I_N * OO_N * 4);
  float*  maskb  = (float*) take((size_t)S_N * I_N * OO_N * 4);
  float*  swbb   = (float*) take((size_t)S_N * I_N * OO_N * 4);
  float*  sderb  = (float*) take((size_t)S_N * I_N * OO_N * 4);
  float*  rmb    = (float*) take((size_t)S_N * I_N * OO_N * 4);
  float*  anum   = (float*) take(256);
  float*  aden   = (float*) take(256);

  prep_feat<<<dim3(256, S_N * I_N), 256, 0, stream>>>(feat, featA);
  prep_maps<<<dim3(17, S_N), 256, 0, stream>>>(bb, label_w, mask_w, spat_w,
                                               labelb, maskb, swbb, lossd);
  prep_w<<<2048, 256, 0, stream>>>(weights, wcur, wB, out);

  for (int t = 0; t < NITER; ++t) {
    gemm_mn<<<dim3(61, S_N), 128, 0, stream>>>(featA, wB, part1);
    scores_resid<<<dim3(17, S_N), 256, 0, stream>>>(part1, labelb, maskb, swbb,
                                                    sderb, rmb, lossd + t, 1);
    reg_loss<<<2048, 256, 0, stream>>>(wcur, freg, lossd + t);
    shift_res<<<dim3(242, S_N), 256, 0, stream>>>(rmb, resT, anum, aden);
    gemm_grad<<<dim3(8, S_N), 128, 0, stream>>>(feat, resT, wcur, freg,
                                                wgrad, wgradB, anum);
    gemm_mn<<<dim3(61, S_N), 128, 0, stream>>>(featA, wgradB, part2);
    sg_reduce<<<dim3(17, S_N), 256, 0, stream>>>(part2, swbb, sderb, aden);
    update_w<<<2048, 256, 0, stream>>>(wgrad, anum, aden, lsl, freg, wcur, wB, out, t);
  }
  // final loss with w_5
  gemm_mn<<<dim3(61, S_N), 128, 0, stream>>>(featA, wB, part1);
  scores_resid<<<dim3(17, S_N), 256, 0, stream>>>(part1, labelb, maskb, swbb,
                                                  sderb, rmb, lossd + NITER, 0);
  reg_loss<<<2048, 256, 0, stream>>>(wcur, freg, lossd + NITER);
}
